// LanguageVisionAttention_37323265802665
// MI455X (gfx1250) — matrix operand order repaired and checked by reading, not compiled, not run on hardware
//
#include <hip/hip_runtime.h>
#include <hip/hip_bf16.h>

// ---------------------------------------------------------------------------
// Types for CDNA5 WMMA (wave32): v16bf A/B fragments, v8f C/D accumulators.
// ---------------------------------------------------------------------------
typedef __bf16 bf16;
typedef __attribute__((ext_vector_type(16))) __bf16 v16bf;
typedef __attribute__((ext_vector_type(8)))  float  v8f;

struct U128 { unsigned long long lo, hi; };

union Frag16 {
  v16bf v;
  bf16  e[16];
  U128  q[2];
};

#define WMMA_BF16(A, B, C) \
  __builtin_amdgcn_wmma_f32_16x16x32_bf16(false, (A), false, (B), (short)0, (C), false, false)

// Problem constants (from reference)
static constexpr int Bb   = 4;
static constexpr int NL   = 1024;
static constexpr int HWC  = 4096;
static constexpr int CL   = 512;
static constexpr int CV   = 512;
static constexpr int CK   = 256;
static constexpr int CVAL = 256;
static constexpr int COUT = 256;
static constexpr int H    = 8;

enum { OUT_ROW_F32 = 0, OUT_ROW_BF16 = 1, OUT_COL_BF16 = 2 };

// ---------------------------------------------------------------------------
// Generic projection GEMM:  out[m,n] = (sum_k A[m,k]*W[n,k] + bias[n])*outScale
// Block tile: 128(M) x 64(N), K-step 32.  8 waves arranged 4(M) x 2(N),
// each wave computes a 32x32 patch = 2x2 WMMA 16x16 tiles.
// A/W are f32 in global; converted to bf16 in LDS; f32 accumulation.
// outScale folds the attention 1/sqrt(Ck) into the Q projection for free.
// ---------------------------------------------------------------------------
template<int MODE>
__global__ __launch_bounds__(256)
void gemm_proj(const float* __restrict__ A, const float* __restrict__ W,
               const float* __restrict__ bias, void* __restrict__ outv,
               int M, int N, int K, long long strideA, long long strideO, int ldo,
               float outScale)
{
  __shared__ __align__(16) bf16 a_lds[128][32];
  __shared__ __align__(16) bf16 w_lds[64][32];

  const int t      = threadIdx.x;
  const int wid    = t >> 5;
  const int lane   = t & 31;
  const int lanelo = lane & 15;
  const int khalf  = lane >> 4;
  const int waveM  = wid & 3;
  const int waveN  = wid >> 2;
  const int b      = blockIdx.z;
  const int mBase  = blockIdx.x * 128;
  const int nBase  = blockIdx.y * 64;

  const float* Ab = A + (long long)b * strideA;

  const v8f vzero = {0.f,0.f,0.f,0.f,0.f,0.f,0.f,0.f};
  v8f acc[2][2];
#pragma unroll
  for (int i = 0; i < 2; ++i)
#pragma unroll
    for (int j = 0; j < 2; ++j) acc[i][j] = vzero;

  for (int k0 = 0; k0 < K; k0 += 32) {
    // --- A tile 128x32 f32 -> bf16 LDS (2 threads per row, 16 f32 each) ---
    {
      const int row = t >> 1, hh = t & 1;
      const float* src = Ab + (size_t)(mBase + row) * K + k0 + hh * 16;
      float4 f0 = ((const float4*)src)[0];
      float4 f1 = ((const float4*)src)[1];
      float4 f2 = ((const float4*)src)[2];
      float4 f3 = ((const float4*)src)[3];
      bf16* d = &a_lds[row][hh * 16];
      d[0]=(bf16)f0.x; d[1]=(bf16)f0.y; d[2]=(bf16)f0.z; d[3]=(bf16)f0.w;
      d[4]=(bf16)f1.x; d[5]=(bf16)f1.y; d[6]=(bf16)f1.z; d[7]=(bf16)f1.w;
      d[8]=(bf16)f2.x; d[9]=(bf16)f2.y; d[10]=(bf16)f2.z; d[11]=(bf16)f2.w;
      d[12]=(bf16)f3.x; d[13]=(bf16)f3.y; d[14]=(bf16)f3.z; d[15]=(bf16)f3.w;
    }
    // --- W tile 64x32 f32 -> bf16 LDS (4 threads per row, 8 f32 each) ---
    {
      const int row = t >> 2, qq = t & 3;
      const float* src = W + (size_t)(nBase + row) * K + k0 + qq * 8;
      float4 f0 = ((const float4*)src)[0];
      float4 f1 = ((const float4*)src)[1];
      bf16* d = &w_lds[row][qq * 8];
      d[0]=(bf16)f0.x; d[1]=(bf16)f0.y; d[2]=(bf16)f0.z; d[3]=(bf16)f0.w;
      d[4]=(bf16)f1.x; d[5]=(bf16)f1.y; d[6]=(bf16)f1.z; d[7]=(bf16)f1.w;
    }
    __syncthreads();

    // A-fragment: lane holds row (lane&15); K runs [khalf*8, +8) and [16+khalf*8, +8)
    Frag16 af[2];
#pragma unroll
    for (int tm = 0; tm < 2; ++tm) {
      const int m = waveM * 32 + tm * 16 + lanelo;
      af[tm].q[0] = *(const U128*)&a_lds[m][khalf * 8];
      af[tm].q[1] = *(const U128*)&a_lds[m][16 + khalf * 8];
    }
    // B-fragment: lane holds col (lane&15); K runs [khalf*8, +8) and [16+khalf*8, +8)
    Frag16 bfr[2];
#pragma unroll
    for (int tn = 0; tn < 2; ++tn) {
      const int n = waveN * 32 + tn * 16 + lanelo;
      bfr[tn].q[0] = *(const U128*)&w_lds[n][khalf * 8];
      bfr[tn].q[1] = *(const U128*)&w_lds[n][16 + khalf * 8];
    }
#pragma unroll
    for (int tm = 0; tm < 2; ++tm)
#pragma unroll
      for (int tn = 0; tn < 2; ++tn)
        acc[tm][tn] = WMMA_BF16(af[tm].v, bfr[tn].v, acc[tm][tn]);
    __syncthreads();
  }

  // Epilogue: C/D layout => lane holds rows khalf*8 + r (r=0..7), col lane&15
#pragma unroll
  for (int tm = 0; tm < 2; ++tm) {
#pragma unroll
    for (int tn = 0; tn < 2; ++tn) {
      const int nG = nBase + waveN * 32 + tn * 16 + lanelo;
      const float bv = bias[nG];
      const int mRow0 = mBase + waveM * 32 + tm * 16 + khalf * 8;
#pragma unroll
      for (int r = 0; r < 8; ++r) {
        const float val = (acc[tm][tn][r] + bv) * outScale;
        if (MODE == OUT_ROW_F32) {
          ((float*)outv)[(long long)b * strideO + (size_t)(mRow0 + r) * ldo + nG] = val;
        } else if (MODE == OUT_ROW_BF16) {
          ((bf16*)outv)[(long long)b * strideO + (size_t)(mRow0 + r) * ldo + nG] = (bf16)val;
        } else { // OUT_COL_BF16 (transposed store, contiguous over r)
          ((bf16*)outv)[(long long)b * strideO + (size_t)nG * ldo + (mRow0 + r)] = (bf16)val;
        }
      }
    }
  }
}

// ---------------------------------------------------------------------------
// Column statistics for InstanceNorm: src is [B, R, 256] (channel-contiguous).
// One block per (b, 256-row chunk); thread = channel; f32 atomics to sums.
// ---------------------------------------------------------------------------
__global__ __launch_bounds__(256)
void colstats(const float* __restrict__ src, float* __restrict__ sum,
              float* __restrict__ sumsq, int R)
{
  const int c = threadIdx.x;
  const int b = blockIdx.y;
  const int r0 = blockIdx.x * 256;
  const float* p = src + ((size_t)b * R + r0) * 256 + c;
  float s = 0.f, ss = 0.f;
  for (int i = 0; i < 256; ++i) {
    const float v = p[(size_t)i * 256];
    s += v; ss += v * v;
  }
  atomicAdd(&sum[b * 256 + c], s);
  atomicAdd(&sumsq[b * 256 + c], ss);
}

// ---------------------------------------------------------------------------
// Apply InstanceNorm (biased var, eps=1e-5): dst = (x-mean)*rsqrt(var+eps)
// src layout [B, R, 256]; OutT = bf16 (for k) or float (final output).
// ---------------------------------------------------------------------------
template<typename OutT>
__global__ __launch_bounds__(256)
void inorm_apply(const float* __restrict__ src, const float* __restrict__ sum,
                 const float* __restrict__ sumsq, OutT* __restrict__ dst, int R)
{
  const size_t idx = (size_t)blockIdx.x * 256 + threadIdx.x;
  const int c = (int)(idx & 255);
  const size_t row = idx >> 8;           // b*R + r
  const int b = (int)(row / (size_t)R);
  const float invR = 1.0f / (float)R;
  const float m = sum[b * 256 + c] * invR;
  const float var = sumsq[b * 256 + c] * invR - m * m;
  const float x = src[idx];
  dst[idx] = (OutT)((x - m) * rsqrtf(var + 1e-5f));
}

// ---------------------------------------------------------------------------
// Flash attention: grid (Nl/128, H, B), 256 threads = 8 waves, wave owns 16
// queries.  dk=dv=32 == bf16 WMMA K.  128-key chunks amortize the cross-lane
// softmax reduction (ds_bpermute butterfly: 4 steps x 8 rows per chunk) and
// the O-rescale over 16 WMMAs:
//   8 WMMA (sim) -> online softmax -> P via per-wave LDS -> 8 WMMA (PV).
// Attention scale 1/sqrt(Ck) is pre-folded into q.
// q: [B,Nl,Ck] bf16   k: [B,HW,Ck] bf16 (normalized)   v: [B,Cval,HW] bf16
// att out: [B,Nl,Cval] f32
// ---------------------------------------------------------------------------
__global__ __launch_bounds__(256)
void attn_kernel(const bf16* __restrict__ q, const bf16* __restrict__ k,
                 const bf16* __restrict__ v, float* __restrict__ att)
{
  __shared__ __align__(16) bf16 p_lds[8][16][128];   // per-wave P staging (32 KB)

  const int t      = threadIdx.x;
  const int w      = t >> 5;
  const int lane   = t & 31;
  const int lanelo = lane & 15;
  const int khalf  = lane >> 4;
  const int b      = blockIdx.z;
  const int h      = blockIdx.y;
  const int l0     = blockIdx.x * 128 + w * 16;

  // q A-fragment (row = lane&15, K runs per ISA A-layout) — register resident
  Frag16 qf;
  {
    const bf16* src = q + ((size_t)b * NL + l0 + lanelo) * CK + h * 32;
    qf.q[0] = *(const U128*)(src + khalf * 8);
    qf.q[1] = *(const U128*)(src + 16 + khalf * 8);
  }

  const v8f vzero = {0.f,0.f,0.f,0.f,0.f,0.f,0.f,0.f};
  v8f o0 = vzero, o1 = vzero;       // O accumulators for d 0..15 / 16..31
  float mrun[8], lrun[8];           // per-row running max / denom (row = khalf*8+r)
#pragma unroll
  for (int r = 0; r < 8; ++r) { mrun[r] = -1e30f; lrun[r] = 0.f; }

  // Loop-invariant fragment base pointers (offsets become clause immediates)
  const bf16* kbase = k + ((size_t)b * HWC + lanelo) * CK + h * 32 + khalf * 8;
  const bf16* vb0   = v + ((size_t)b * CVAL + h * 32 + lanelo) * HWC + khalf * 16;
  const bf16* vb1   = v + ((size_t)b * CVAL + h * 32 + 16 + lanelo) * HWC + khalf * 16;

  for (int nc = 0; nc < HWC; nc += 128) {
    // ---- sim: 8 WMMAs over 128 keys (scale already folded into q) ----
    v8f s[8];
#pragma unroll
    for (int kt = 0; kt < 8; ++kt) {
      Frag16 kf;
      const bf16* src = kbase + (size_t)(nc + kt * 16) * CK;
      kf.q[0] = *(const U128*)src;
      kf.q[1] = *(const U128*)(src + 16);
      s[kt] = WMMA_BF16(qf.v, kf.v, vzero);
    }

    // ---- online softmax: rows 0..7 in lanes 0-15, 8..15 in lanes 16-31.
    // 16 lanes of a half hold one row, so xor masks 1,2,4,8 reduce the row.
#pragma unroll
    for (int r = 0; r < 8; ++r) {
      float a[8];
#pragma unroll
      for (int kt = 0; kt < 8; ++kt) a[kt] = s[kt][r];
      float cm = fmaxf(fmaxf(fmaxf(a[0], a[1]), fmaxf(a[2], a[3])),
                       fmaxf(fmaxf(a[4], a[5]), fmaxf(a[6], a[7])));
      cm = fmaxf(cm, __shfl_xor(cm, 1, 32));
      cm = fmaxf(cm, __shfl_xor(cm, 2, 32));
      cm = fmaxf(cm, __shfl_xor(cm, 4, 32));
      cm = fmaxf(cm, __shfl_xor(cm, 8, 32));
      const float nm = fmaxf(mrun[r], cm);
      const float alpha = __expf(mrun[r] - nm);
      float psum = 0.f;
      bf16* pr = &p_lds[w][khalf * 8 + r][lanelo];
#pragma unroll
      for (int kt = 0; kt < 8; ++kt) {
        const float p = __expf(a[kt] - nm);
        psum += p;
        pr[kt * 16] = (bf16)p;                       // D-layout -> LDS
      }
      mrun[r] = nm;
      lrun[r] = lrun[r] * alpha + psum;              // per-lane partial
      o0[r] *= alpha;
      o1[r] *= alpha;
    }
    __builtin_amdgcn_wave_barrier();   // DS pipe is in-order per wave

    // ---- PV: 4 K-chunks of 32 keys; P A-frags from LDS, V B-frags global ----
#pragma unroll
    for (int kt2 = 0; kt2 < 4; ++kt2) {
      Frag16 pf;
      pf.q[0] = *(const U128*)&p_lds[w][lanelo][kt2 * 32 + khalf * 16];
      pf.q[1] = *(const U128*)&p_lds[w][lanelo][kt2 * 32 + khalf * 16 + 8];

      Frag16 vf0, vf1;
      const bf16* s0p = vb0 + nc + kt2 * 32;
      const bf16* s1p = vb1 + nc + kt2 * 32;
      vf0.q[0] = *(const U128*)s0p; vf0.q[1] = *(const U128*)(s0p + 8);
      vf1.q[0] = *(const U128*)s1p; vf1.q[1] = *(const U128*)(s1p + 8);

      o0 = WMMA_BF16(pf.v, vf0.v, o0);
      o1 = WMMA_BF16(pf.v, vf1.v, o1);
    }
    __builtin_amdgcn_wave_barrier();
  }

  // Finalize: reduce per-lane denom partials across the 16-lane row group.
#pragma unroll
  for (int r = 0; r < 8; ++r) {
    float s = lrun[r];
    s += __shfl_xor(s, 1, 32);
    s += __shfl_xor(s, 2, 32);
    s += __shfl_xor(s, 4, 32);
    s += __shfl_xor(s, 8, 32);
    const float inv = __builtin_amdgcn_rcpf(s);
    const size_t rowoff = ((size_t)b * NL + l0 + khalf * 8 + r) * CVAL + h * 32;
    att[rowoff + lanelo]      = o0[r] * inv;
    att[rowoff + 16 + lanelo] = o1[r] * inv;
  }
}

// ---------------------------------------------------------------------------
// Host-side launch
// ---------------------------------------------------------------------------
extern "C" void kernel_launch(void* const* d_in, const int* in_sizes, int n_in,
                              void* d_out, int out_size, void* d_ws, size_t ws_size,
                              hipStream_t stream) {
  (void)in_sizes; (void)n_in; (void)out_size; (void)ws_size;
  const float* l  = (const float*)d_in[0];   // [B, Nl, Cl]
  const float* x  = (const float*)d_in[1];   // [B, HW, Cv]
  const float* Wq = (const float*)d_in[2];   // [Ck, Cl]
  const float* bq = (const float*)d_in[3];
  const float* Wk = (const float*)d_in[4];   // [Ck, Cv]
  const float* bk = (const float*)d_in[5];
  const float* Wv = (const float*)d_in[6];   // [Cval, Cv]
  const float* bv = (const float*)d_in[7];
  const float* Ww = (const float*)d_in[8];   // [Cout, Cval]
  const float* bw = (const float*)d_in[9];

  // Workspace carve-out (all sizes 256B-aligned)
  char* ws = (char*)d_ws;
  size_t off = 0;
  bf16*  q_bf   = (bf16*)(ws + off);  off += (size_t)Bb * NL  * CK   * 2;  // 2 MB
  float* k_f32  = (float*)(ws + off); off += (size_t)Bb * HWC * CK   * 4;  // 16 MB
  bf16*  k_bf   = (bf16*)(ws + off);  off += (size_t)Bb * HWC * CK   * 2;  // 8 MB
  bf16*  v_bf   = (bf16*)(ws + off);  off += (size_t)Bb * CVAL* HWC  * 2;  // 8 MB
  float* att_f  = (float*)(ws + off); off += (size_t)Bb * NL  * CVAL * 4;  // 4 MB
  float* proj_f = (float*)(ws + off); off += (size_t)Bb * NL  * COUT * 4;  // 4 MB
  float* stats  = (float*)(ws + off);                                      // 4 x 4KB
  float* ksum   = stats;
  float* ksumsq = stats + Bb * 256;
  float* psum   = stats + 2 * Bb * 256;
  float* psumsq = stats + 3 * Bb * 256;

  hipMemsetAsync(stats, 0, (size_t)4 * Bb * 256 * sizeof(float), stream);

  // Q = (l @ Wq^T + bq) * Ck^-0.5  -> bf16 [B, Nl, Ck]  (scale folded in)
  gemm_proj<OUT_ROW_BF16><<<dim3(NL / 128, CK / 64, Bb), 256, 0, stream>>>(
      l, Wq, bq, q_bf, NL, CK, CL, (long long)NL * CL, (long long)NL * CK, CK, 0.0625f);
  // K(raw) = x @ Wk^T + bk -> f32 [B, HW, Ck]
  gemm_proj<OUT_ROW_F32><<<dim3(HWC / 128, CK / 64, Bb), 256, 0, stream>>>(
      x, Wk, bk, k_f32, HWC, CK, CV, (long long)HWC * CV, (long long)HWC * CK, CK, 1.0f);
  // V = x @ Wv^T + bv -> bf16 transposed [B, Cval, HW]
  gemm_proj<OUT_COL_BF16><<<dim3(HWC / 128, CVAL / 64, Bb), 256, 0, stream>>>(
      x, Wv, bv, v_bf, HWC, CVAL, CV, (long long)HWC * CV, (long long)CVAL * HWC, HWC, 1.0f);

  // InstanceNorm(k) over HW per (b,c): stats then normalize -> bf16
  colstats<<<dim3(HWC / 256, Bb), 256, 0, stream>>>(k_f32, ksum, ksumsq, HWC);
  inorm_apply<bf16><<<dim3((Bb * HWC * 256) / 256), 256, 0, stream>>>(
      k_f32, ksum, ksumsq, k_bf, HWC);

  // Flash attention -> f32 [B, Nl, Cval]
  attn_kernel<<<dim3(NL / 128, H, Bb), 256, 0, stream>>>(q_bf, k_bf, v_bf, att_f);

  // Output projection: att @ Ww^T + bw -> f32 [B, Nl, Cout]
  gemm_proj<OUT_ROW_F32><<<dim3(NL / 128, COUT / 64, Bb), 256, 0, stream>>>(
      att_f, Ww, bw, proj_f, NL, COUT, CVAL, (long long)NL * CVAL, (long long)NL * COUT,
      COUT, 1.0f);

  // InstanceNorm over Nl per (b,o); output layout already [B, Nl, Cout]
  colstats<<<dim3(NL / 256, Bb), 256, 0, stream>>>(proj_f, psum, psumsq, NL);
  inorm_apply<float><<<dim3((Bb * NL * 256) / 256), 256, 0, stream>>>(
      proj_f, psum, psumsq, (float*)d_out, NL);
}
